// ProbAttention_9947144258110
// MI455X (gfx1250) — matrix operand order repaired and checked by reading, not compiled, not run on hardware
//
#include <hip/hip_runtime.h>
#include <hip/hip_bf16.h>
#include <math.h>

#define B_  2
#define LQ  2048
#define LK  2048
#define DM  1024
#define H_  16
#define DK  64
#define DV  64
#define UK  38
#define UQ  38

typedef __attribute__((ext_vector_type(16))) _Float16 v16h;
typedef __attribute__((ext_vector_type(8)))  _Float16 v8h;
typedef __attribute__((ext_vector_type(8)))  float    v8f;

// ---------------------------------------------------------------------------
// GEMM: C[4096 x 1024] = A[4096 x 1024] * W[1024 x 1024]
// f32 in global, f16 in LDS (converted in flight), f32 WMMA accumulation.
// MODE 0: store permuted to (B,H,L,64) layout (projections)
// MODE 1: store row-major + residual add (final fc projection)
// Block: 256 thr (8 waves). Tile: BM=128, BN=64, BK=32. Wave: 32x32 (2x2 WMMA).
// Double-buffered LDS: prefetch next K-tile into regs while WMMA runs.
// ---------------------------------------------------------------------------
#define BM 128
#define BN 64
#define BK 32
#define KSTEPS (DM / BK)
#define ASTR 40   // halves per A row (32 + pad), 80B => 16B-aligned rows
#define BSTR 40   // halves per Bt row

template <int MODE>
__global__ __launch_bounds__(256)
void gemm_f16_wmma(const float* __restrict__ A, const float* __restrict__ W,
                   float* __restrict__ Out, const float* __restrict__ Res) {
  __shared__ __align__(16) _Float16 As[2][BM * ASTR];
  __shared__ __align__(16) _Float16 Bt[2][BN * BSTR];

  const int t    = threadIdx.x;
  const int wave = t >> 5, lane = t & 31;
  const int half = lane >> 4, l15 = lane & 15;
  const int wm = wave >> 1, wn = wave & 1;
  const int m0 = blockIdx.y * BM;
  const int n0 = blockIdx.x * BN;

  const v8f vz = {0.f,0.f,0.f,0.f,0.f,0.f,0.f,0.f};
  v8f acc[2][2] = {{vz, vz}, {vz, vz}};

  // per-thread load coordinates (compile-time strides -> shift/mask only)
  float4 ra[4], rb[2];

  auto loadTile = [&](int k0) {
#pragma unroll
    for (int i = 0; i < 4; ++i) {                 // A tile: 128x32 f32
      const int e4 = t + i * 256;
      const int r = e4 >> 3, c4 = (e4 & 7) << 2;
      ra[i] = *(const float4*)(A + (((size_t)(m0 + r)) << 10) + k0 + c4);
    }
#pragma unroll
    for (int i = 0; i < 2; ++i) {                 // B tile: 32x64 f32
      const int e4 = t + i * 256;
      const int k = e4 >> 4, n4 = (e4 & 15) << 2;
      rb[i] = *(const float4*)(W + (((size_t)(k0 + k)) << 10) + n0 + n4);
    }
  };
  auto stageTile = [&](int buf) {
#pragma unroll
    for (int i = 0; i < 4; ++i) {
      const int e4 = t + i * 256;
      const int r = e4 >> 3, c4 = (e4 & 7) << 2;
      _Float16* p = &As[buf][r * ASTR + c4];
      p[0] = (_Float16)ra[i].x; p[1] = (_Float16)ra[i].y;
      p[2] = (_Float16)ra[i].z; p[3] = (_Float16)ra[i].w;
    }
#pragma unroll
    for (int i = 0; i < 2; ++i) {
      const int e4 = t + i * 256;
      const int k = e4 >> 4, n4 = (e4 & 15) << 2;
      Bt[buf][(n4 + 0) * BSTR + k] = (_Float16)rb[i].x;   // transposed: Bt[n][k]
      Bt[buf][(n4 + 1) * BSTR + k] = (_Float16)rb[i].y;
      Bt[buf][(n4 + 2) * BSTR + k] = (_Float16)rb[i].z;
      Bt[buf][(n4 + 3) * BSTR + k] = (_Float16)rb[i].w;
    }
  };

  loadTile(0);
  stageTile(0);
  __syncthreads();

  for (int kt = 0; kt < KSTEPS; ++kt) {
    const int cur = kt & 1;
    if (kt + 1 < KSTEPS) loadTile((kt + 1) * BK);   // prefetch overlaps WMMA

    // ---- fragments per ISA 16-bit layouts ----
    v16h afrag[2], bfrag[2];
#pragma unroll
    for (int i = 0; i < 2; ++i) {
      // A 16x32: lane half h: K in [8h,8h+8) then [16+8h,16+8h+8)
      const _Float16* p = &As[cur][(wm * 32 + i * 16 + l15) * ASTR + 8 * half];
      union { v16h v; v8h h[2]; } u;
      u.h[0] = *(const v8h*)p;
      u.h[1] = *(const v8h*)(p + 16);
      afrag[i] = u.v;
    }
#pragma unroll
    for (int j = 0; j < 2; ++j) {
      // B 32x16: lane half h: K in [8h,8h+8) then [16+8h,16+8h+8), column n = l15
      const _Float16* p = &Bt[cur][(wn * 32 + j * 16 + l15) * BSTR + 8 * half];
      union { v16h v; v8h h[2]; } u;
      u.h[0] = *(const v8h*)p;
      u.h[1] = *(const v8h*)(p + 16);
      bfrag[j] = u.v;
    }
#pragma unroll
    for (int i = 0; i < 2; ++i)
#pragma unroll
      for (int j = 0; j < 2; ++j)
        acc[i][j] = __builtin_amdgcn_wmma_f32_16x16x32_f16(
            false, afrag[i], false, bfrag[j], (short)0, acc[i][j], false, false);

    if (kt + 1 < KSTEPS) {
      stageTile(cur ^ 1);
      __syncthreads();
    }
  }

  // ---- epilogue: C/D layout: lane(h=lane>>4, n=l15), vgpr r -> M = 8h+r ----
#pragma unroll
  for (int i = 0; i < 2; ++i)
#pragma unroll
    for (int j = 0; j < 2; ++j) {
      const int R0 = m0 + wm * 32 + i * 16 + half * 8;
      const int C  = n0 + wn * 32 + j * 16 + l15;
      if (MODE == 0) {
        const int h = C >> 6, d = C & 63;
#pragma unroll
        for (int r = 0; r < 8; ++r) {
          const int R = R0 + r;
          const int b = R >> 11, lr = R & (LQ - 1);
          Out[(((((size_t)b * H_ + h) << 11) + lr) << 6) + d] = acc[i][j][r];
        }
      } else {
#pragma unroll
        for (int r = 0; r < 8; ++r) {
          const size_t off = (((size_t)(R0 + r)) << 10) + C;
          Out[off] = acc[i][j][r] + Res[off];
        }
      }
    }
}

// ---------------------------------------------------------------------------
// Sparse-sampled QK metric: M[b,h,q] = max_s(dot) - mean_s(dot), one wave/query
// ---------------------------------------------------------------------------
__global__ __launch_bounds__(256)
void qk_sample_metric(const float* __restrict__ Q, const float* __restrict__ K,
                      const int* __restrict__ samp, float* __restrict__ Mout) {
  const int gw   = blockIdx.x * 8 + (threadIdx.x >> 5);  // (b*H+h)*LQ + q
  const int lane = threadIdx.x & 31;
  const int bh   = gw >> 11;
  const int q    = gw & (LQ - 1);
  const float* qrow = Q + ((size_t)gw << 6);

  float lmax = -3.4e38f, lsum = 0.f;
  for (int s = lane; s < UK; s += 32) {
    const int kidx = samp[q * UK + s];
    const float* krow = K + ((((size_t)bh << 11) + kidx) << 6);
    float dot = 0.f;
#pragma unroll
    for (int d = 0; d < 64; d += 4) {
      const float4 qv = *(const float4*)(qrow + d);
      const float4 kv = *(const float4*)(krow + d);
      dot += qv.x * kv.x + qv.y * kv.y + qv.z * kv.z + qv.w * kv.w;
    }
    lmax = fmaxf(lmax, dot);
    lsum += dot;
  }
#pragma unroll
  for (int off = 16; off > 0; off >>= 1) {
    lmax = fmaxf(lmax, __shfl_xor(lmax, off, 32));
    lsum += __shfl_xor(lsum, off, 32);
  }
  if (lane == 0) Mout[gw] = lmax - lsum * (1.0f / UK);
}

// ---------------------------------------------------------------------------
// Top-UQ indices per (b,h): iterative block argmax over 2048 metric values
// ---------------------------------------------------------------------------
__global__ __launch_bounds__(256)
void topk_kernel(const float* __restrict__ Mbuf, int* __restrict__ idxOut) {
  __shared__ float sm[LQ];
  __shared__ float bv[256];
  __shared__ int   bi[256];
  const int bh = blockIdx.x, t = threadIdx.x;
  for (int i = t; i < LQ; i += 256) sm[i] = Mbuf[bh * LQ + i];
  __syncthreads();
  for (int it = 0; it < UQ; ++it) {
    float best = -3.4e38f; int besti = 0;
    for (int i = t; i < LQ; i += 256) {
      const float v = sm[i];
      if (v > best) { best = v; besti = i; }
    }
    bv[t] = best; bi[t] = besti;
    __syncthreads();
    for (int s = 128; s > 0; s >>= 1) {
      if (t < s) {
        if (bv[t + s] > bv[t] ||
            (bv[t + s] == bv[t] && bi[t + s] < bi[t])) {
          bv[t] = bv[t + s]; bi[t] = bi[t + s];
        }
      }
      __syncthreads();
    }
    if (t == 0) { idxOut[bh * UQ + it] = bi[0]; sm[bi[0]] = -3.4e38f; }
    __syncthreads();
  }
}

// ---------------------------------------------------------------------------
// V mean over L per (b,h,d)
// ---------------------------------------------------------------------------
__global__ void v_mean_kernel(const float* __restrict__ V, float* __restrict__ Vm) {
  const int bh = blockIdx.x, d = threadIdx.x;
  const float* base = V + (((size_t)bh) << 17) + d;   // bh*2048*64
  float s = 0.f;
  for (int l = 0; l < LK; ++l) s += base[(size_t)l << 6];
  Vm[bh * 64 + d] = s * (1.0f / LK);
}

// ---------------------------------------------------------------------------
// context[b, l, h*64+d] = Vmean[b,h,d] (broadcast fill)
// ---------------------------------------------------------------------------
__global__ void fill_context_kernel(const float* __restrict__ Vm,
                                    float* __restrict__ ctx) {
  const size_t e = (size_t)blockIdx.x * 256 + threadIdx.x;   // B*LQ*1024 elems
  const int c = (int)(e & 1023);
  const int b = (int)(e >> 21);                              // LQ*1024 = 2^21
  ctx[e] = Vm[b * 1024 + c];
}

// ---------------------------------------------------------------------------
// Full attention for the UQ selected queries; one block per (b,h,u)
// ---------------------------------------------------------------------------
__global__ __launch_bounds__(256)
void sparse_attn_kernel(const float* __restrict__ Q, const float* __restrict__ K,
                        const float* __restrict__ V, const int* __restrict__ idx,
                        float* __restrict__ ctx) {
  __shared__ float sQ[64];
  __shared__ float p[LK];
  __shared__ float red[256];
  __shared__ float part[4][64];
  const int t  = threadIdx.x;
  const int g  = blockIdx.x;
  const int bh = g / UQ, u = g % UQ;
  const int b  = bh >> 4, h = bh & 15;
  const int q  = idx[bh * UQ + u];

  if (t < 64) sQ[t] = Q[((((size_t)bh << 11) + q) << 6) + t];
  __syncthreads();

  float lmax = -3.4e38f;
  float sloc[8];
#pragma unroll
  for (int i = 0; i < 8; ++i) {
    const int k = t + i * 256;
    const float* kr = K + ((((size_t)bh << 11) + k) << 6);
    float dot = 0.f;
#pragma unroll
    for (int d = 0; d < 64; d += 4) {
      const float4 kv = *(const float4*)(kr + d);
      dot += sQ[d] * kv.x + sQ[d+1] * kv.y + sQ[d+2] * kv.z + sQ[d+3] * kv.w;
    }
    sloc[i] = dot * 0.125f;           // 1/sqrt(64)
    lmax = fmaxf(lmax, sloc[i]);
  }
  red[t] = lmax; __syncthreads();
  for (int s = 128; s > 0; s >>= 1) {
    if (t < s) red[t] = fmaxf(red[t], red[t + s]);
    __syncthreads();
  }
  const float gmax = red[0];
  __syncthreads();

  float lsum = 0.f;
#pragma unroll
  for (int i = 0; i < 8; ++i) {
    const float e = __expf(sloc[i] - gmax);
    p[t + i * 256] = e;
    lsum += e;
  }
  red[t] = lsum; __syncthreads();
  for (int s = 128; s > 0; s >>= 1) {
    if (t < s) red[t] += red[t + s];
    __syncthreads();
  }
  const float inv = 1.0f / red[0];
  __syncthreads();

  const int d = t & 63, ch = t >> 6;
  const float* vb = V + ((((size_t)bh << 11) + ch * 512) << 6) + d;
  float o = 0.f;
  for (int k = 0; k < 512; ++k) o += p[ch * 512 + k] * vb[(size_t)k << 6];
  part[ch][d] = o;
  __syncthreads();
  if (t < 64) {
    const float val = (part[0][t] + part[1][t] + part[2][t] + part[3][t]) * inv;
    ctx[((((size_t)b << 11) + q) << 10) + h * 64 + t] = val;
  }
}

// ---------------------------------------------------------------------------
// Row LayerNorm over 1024 cols, one block per row, 4 elems/thread
// ---------------------------------------------------------------------------
__global__ __launch_bounds__(256)
void layernorm_kernel(const float* __restrict__ X, const float* __restrict__ gamma,
                      const float* __restrict__ beta, float* __restrict__ out) {
  __shared__ float red[256];
  const int r = blockIdx.x, t = threadIdx.x;
  const float* row = X + ((size_t)r << 10);
  const float4 x = *(const float4*)(row + t * 4);
  red[t] = x.x + x.y + x.z + x.w;
  __syncthreads();
  for (int s = 128; s > 0; s >>= 1) { if (t < s) red[t] += red[t + s]; __syncthreads(); }
  const float mu = red[0] * (1.0f / DM);
  __syncthreads();
  const float4 dx = { x.x - mu, x.y - mu, x.z - mu, x.w - mu };
  red[t] = dx.x*dx.x + dx.y*dx.y + dx.z*dx.z + dx.w*dx.w;
  __syncthreads();
  for (int s = 128; s > 0; s >>= 1) { if (t < s) red[t] += red[t + s]; __syncthreads(); }
  const float rstd = rsqrtf(red[0] * (1.0f / DM) + 1e-5f);
  const float4 g   = *(const float4*)(gamma + t * 4);
  const float4 bt  = *(const float4*)(beta  + t * 4);
  float4 o = { g.x * dx.x * rstd + bt.x, g.y * dx.y * rstd + bt.y,
               g.z * dx.z * rstd + bt.z, g.w * dx.w * rstd + bt.w };
  *(float4*)(out + ((size_t)r << 10) + t * 4) = o;
}

// ---------------------------------------------------------------------------
extern "C" void kernel_launch(void* const* d_in, const int* in_sizes, int n_in,
                              void* d_out, int out_size, void* d_ws, size_t ws_size,
                              hipStream_t stream) {
  const float* inQ   = (const float*)d_in[0];
  const float* inK   = (const float*)d_in[1];
  const float* inV   = (const float*)d_in[2];
  const float* WQ    = (const float*)d_in[3];
  const float* WK    = (const float*)d_in[4];
  const float* WV    = (const float*)d_in[5];
  const float* Wfc   = (const float*)d_in[6];
  const float* gamma = (const float*)d_in[7];
  const float* beta  = (const float*)d_in[8];
  const int*   samp  = (const int*)d_in[9];

  char* ws = (char*)d_ws;
  const size_t SZ_BHLD = (size_t)B_ * H_ * LQ * DK * sizeof(float);   // 16 MB
  float* Qb   = (float*)ws; ws += SZ_BHLD;
  float* Kb   = (float*)ws; ws += SZ_BHLD;
  float* Vb   = (float*)ws; ws += SZ_BHLD;
  float* ctx  = (float*)ws; ws += (size_t)B_ * LQ * H_ * DV * sizeof(float);
  float* opre = (float*)ws; ws += (size_t)B_ * LQ * DM * sizeof(float);
  float* Mb   = (float*)ws; ws += (size_t)B_ * H_ * LQ * sizeof(float);
  float* Vm   = (float*)ws; ws += (size_t)B_ * H_ * DV * sizeof(float);
  int*   idxb = (int*)ws;   ws += 4096;

  dim3 gg(DM / BN, (B_ * LQ) / BM);   // (16, 32)
  gemm_f16_wmma<0><<<gg, 256, 0, stream>>>(inQ, WQ, Qb, nullptr);
  gemm_f16_wmma<0><<<gg, 256, 0, stream>>>(inK, WK, Kb, nullptr);
  gemm_f16_wmma<0><<<gg, 256, 0, stream>>>(inV, WV, Vb, nullptr);

  qk_sample_metric<<<(B_ * H_ * LQ) / 8, 256, 0, stream>>>(Qb, Kb, samp, Mb);
  topk_kernel<<<B_ * H_, 256, 0, stream>>>(Mb, idxb);
  v_mean_kernel<<<B_ * H_, 64, 0, stream>>>(Vb, Vm);
  fill_context_kernel<<<(B_ * LQ * DM) / 256, 256, 0, stream>>>(Vm, ctx);
  sparse_attn_kernel<<<B_ * H_ * UQ, 256, 0, stream>>>(Qb, Kb, Vb, idxb, ctx);

  gemm_f16_wmma<1><<<gg, 256, 0, stream>>>(ctx, Wfc, opre, inQ);
  layernorm_kernel<<<B_ * LQ, 256, 0, stream>>>(opre, gamma, beta, (float*)d_out);
}
